// HARSpikingNet_41747082117859
// MI455X (gfx1250) — matrix operand order repaired and checked by reading, not compiled, not run on hardware
//
#include <hip/hip_runtime.h>

#define T_STEPS 128
#define B_SZ 512
#define D_SZ 512
#define H_SZ 2048
#define C_SZ 18

#define BM 32            // batch tile rows per WG
#define BN 64            // hidden tile cols per WG
#define LDS_K (D_SZ + 8) // 520 bf16 per row -> 260 dwords, balanced 2-way b128 banks
#define THREADS 256

typedef __attribute__((ext_vector_type(16))) __bf16 v16bf;
typedef __attribute__((ext_vector_type(8)))  float  v8f;

static __device__ __forceinline__ unsigned short f2bf(float f) {
  union { float f; unsigned u; } x; x.f = f;
  unsigned r = x.u + 0x7FFFu + ((x.u >> 16) & 1u);   // round-to-nearest-even
  return (unsigned short)(r >> 16);
}

// Fused: h_t = x_t @ W_in^T + b_in ; LIF scan kept in registers; emit z at t=T-1.
__global__ __launch_bounds__(THREADS)
void lif_scan_kernel(const float* __restrict__ x,
                     const float* __restrict__ W_in,
                     const float* __restrict__ b_in,
                     unsigned short* __restrict__ z_ws) {
  extern __shared__ unsigned short smem[];
  unsigned short* Wl = smem;                 // BN x LDS_K  (bf16, resident all T)
  unsigned short* Xl = smem + BN * LDS_K;    // BM x LDS_K  (bf16, refilled per t)

  const int tid  = threadIdx.x;
  const int lane = tid & 31;
  const int w    = tid >> 5;        // 8 waves
  const int r    = lane & 15;
  const int hi   = lane >> 4;
  const int mA   = (w & 1) * 16;    // wave's 16-row strip within BM
  const int nB   = (w >> 1) * 16;   // wave's 16-col strip within BN

  const int b_base = blockIdx.x * BM;
  const int h_base = blockIdx.y * BN;

  // --- W_in tile (BN x D) f32 -> bf16 LDS, once ---
  for (int j = tid; j < BN * D_SZ / 4; j += THREADS) {
    int row = j >> 7;                // 128 float4 chunks per 512-wide row
    int c4  = j & 127;
    float4 v = *(const float4*)(W_in + (size_t)(h_base + row) * D_SZ + c4 * 4);
    unsigned short* p = Wl + row * LDS_K + c4 * 4;
    p[0] = f2bf(v.x); p[1] = f2bf(v.y); p[2] = f2bf(v.z); p[3] = f2bf(v.w);
  }

  const float bias = b_in[h_base + nB + r];

  v8f vstate = {0.f,0.f,0.f,0.f,0.f,0.f,0.f,0.f};
  v8f zlast  = {0.f,0.f,0.f,0.f,0.f,0.f,0.f,0.f};

  // register prefetch of x tile for t=0 (16 float4 per thread = BM*D/THREADS)
  float4 xr[16];
  {
    const float* xt = x + (size_t)b_base * D_SZ;
#pragma unroll
    for (int j = 0; j < 16; ++j) {
      int ch = tid + j * THREADS;
      int row = ch >> 7, c4 = ch & 127;
      xr[j] = *(const float4*)(xt + (size_t)row * D_SZ + c4 * 4);
    }
  }

  for (int t = 0; t < T_STEPS; ++t) {
    __syncthreads();                         // prior WMMAs done reading Xl
    // commit prefetched x_t to LDS as bf16
#pragma unroll
    for (int j = 0; j < 16; ++j) {
      int ch = tid + j * THREADS;
      int row = ch >> 7, c4 = ch & 127;
      unsigned short* p = Xl + row * LDS_K + c4 * 4;
      float4 v = xr[j];
      p[0] = f2bf(v.x); p[1] = f2bf(v.y); p[2] = f2bf(v.z); p[3] = f2bf(v.w);
    }
    __syncthreads();

    // issue global loads for t+1; latency hidden behind the WMMA loop
    {
      int tn = (t < T_STEPS - 1) ? (t + 1) : t;
      const float* xt = x + ((size_t)tn * B_SZ + b_base) * D_SZ;
#pragma unroll
      for (int j = 0; j < 16; ++j) {
        int ch = tid + j * THREADS;
        int row = ch >> 7, c4 = ch & 127;
        xr[j] = *(const float4*)(xt + (size_t)row * D_SZ + c4 * 4);
      }
    }

    // h = bias + X_tile(16x512) * W_tile^T(512x16), K accumulated in f32
    v8f acc;
#pragma unroll
    for (int i = 0; i < 8; ++i) acc[i] = bias;

    const unsigned short* Abase = Xl + (mA + r) * LDS_K + hi * 8;
    const unsigned short* Bbase = Wl + (nB + r) * LDS_K + hi * 8;
#pragma unroll
    for (int kk = 0; kk < 16; ++kk) {
      const int k0 = kk * 32;
      v16bf a, b;
      uint4* ap = (uint4*)&a;
      ap[0] = *(const uint4*)(Abase + k0);        // K = hi*8 .. hi*8+7
      ap[1] = *(const uint4*)(Abase + k0 + 16);   // K = 16+hi*8 ..
      uint4* bp = (uint4*)&b;
      bp[0] = *(const uint4*)(Bbase + k0);        // K = hi*8 .. hi*8+7
      bp[1] = *(const uint4*)(Bbase + k0 + 16);   // K = 16+hi*8 ..
      acc = __builtin_amdgcn_wmma_f32_16x16x32_bf16(
          false, a, false, b, (short)0, acc, false, false);
    }

    // LIF: v = 0.9 v + h ; z = H(v-1) ; v -= z
#pragma unroll
    for (int i = 0; i < 8; ++i) {
      float v = 0.9f * vstate[i] + acc[i];
      float z = (v > 1.0f) ? 1.0f : 0.0f;
      vstate[i] = v - z;
      zlast[i]  = z;
    }
  }

  // write z_{T-1} (bf16) to workspace; C layout: VGPR i -> M = i + 8*hi, N = lane&15
  {
    const int n = h_base + nB + r;
#pragma unroll
    for (int i = 0; i < 8; ++i) {
      int m = b_base + mA + hi * 8 + i;
      z_ws[(size_t)m * H_SZ + n] = f2bf(zlast[i]);
    }
  }
}

// logits = z_last @ W_out^T + b_out  (512x18, negligible)
__global__ __launch_bounds__(256)
void out_proj_kernel(const unsigned short* __restrict__ z_ws,
                     const float* __restrict__ W_out,
                     const float* __restrict__ b_out,
                     float* __restrict__ out) {
  int gid = blockIdx.x * blockDim.x + threadIdx.x;
  if (gid >= B_SZ * C_SZ) return;
  int c = gid % C_SZ;
  int b = gid / C_SZ;
  const unsigned short* zr = z_ws + (size_t)b * H_SZ;
  const float* wr = W_out + (size_t)c * H_SZ;
  float acc = b_out[c];
  for (int h = 0; h < H_SZ; h += 8) {
    uint4 zp = *(const uint4*)(zr + h);
    const unsigned* zu = (const unsigned*)&zp;
#pragma unroll
    for (int k = 0; k < 4; ++k) {
      unsigned u = zu[k];
      float z0 = __uint_as_float(u << 16);
      float z1 = __uint_as_float(u & 0xFFFF0000u);
      acc += z0 * wr[h + 2 * k] + z1 * wr[h + 2 * k + 1];
    }
  }
  out[b * C_SZ + c] = acc;
}

extern "C" void kernel_launch(void* const* d_in, const int* in_sizes, int n_in,
                              void* d_out, int out_size, void* d_ws, size_t ws_size,
                              hipStream_t stream) {
  const float* x     = (const float*)d_in[0];
  const float* W_in  = (const float*)d_in[1];
  const float* b_in  = (const float*)d_in[2];
  const float* W_out = (const float*)d_in[3];
  const float* b_out = (const float*)d_in[4];
  unsigned short* z_ws = (unsigned short*)d_ws;   // 512*2048 bf16 = 2 MB
  float* out = (float*)d_out;

  dim3 grid1(B_SZ / BM, H_SZ / BN);               // 16 x 32 = 512 WGs
  size_t smem = (size_t)(BM + BN) * LDS_K * sizeof(unsigned short); // 99,840 B
  lif_scan_kernel<<<grid1, THREADS, smem, stream>>>(x, W_in, b_in, z_ws);

  int total = B_SZ * C_SZ;
  out_proj_kernel<<<(total + 255) / 256, 256, 0, stream>>>(z_ws, W_out, b_out, out);
}
